// TerrainGuidedAttention_29016799052464
// MI455X (gfx1250) — matrix operand order repaired and checked by reading, not compiled, not run on hardware
//
#include <hip/hip_runtime.h>


typedef __attribute__((ext_vector_type(16))) __bf16 v16bf;
typedef __attribute__((ext_vector_type(8)))  float  v8f;

#define BATCH 4
#define NTOK  4096
#define CCH   256
#define DQK   32
#define QKW   64
#define QKVW  320
#define KVT   64

__device__ __forceinline__ __bf16 f2bf(float f) {
  union { float f; unsigned u; } v; v.f = f;
  unsigned r = v.u + 0x7FFFu + ((v.u >> 16) & 1u);
  unsigned short h = (unsigned short)(r >> 16);
  __bf16 o; __builtin_memcpy(&o, &h, 2); return o;
}

__device__ __forceinline__ int a_kpair(int r, int half) {
  return 16 * half + 2 * r;
}

__device__ __forceinline__ float rmax16(float v) {
  v = fmaxf(v, __shfl_xor(v, 1, 16));
  v = fmaxf(v, __shfl_xor(v, 2, 16));
  v = fmaxf(v, __shfl_xor(v, 4, 16));
  v = fmaxf(v, __shfl_xor(v, 8, 16));
  return v;
}
__device__ __forceinline__ float rsum16(float v) {
  v += __shfl_xor(v, 1, 16);
  v += __shfl_xor(v, 2, 16);
  v += __shfl_xor(v, 4, 16);
  v += __shfl_xor(v, 8, 16);
  return v;
}

__global__ void k_tfeat(const float* __restrict__ terrain,
                        const float* __restrict__ Wt,
                        const float* __restrict__ bt,
                        float* __restrict__ tfeat) {
  int b = blockIdx.x;
  int c = threadIdx.x;
  float s = bt[c];
  #pragma unroll 8
  for (int t = 0; t < 64; ++t) s += terrain[b * 64 + t] * Wt[t * CCH + c];
  tfeat[b * CCH + c] = s;
}

__global__ void k_pack(const float* __restrict__ Wq, const float* __restrict__ Wk,
                       const float* __restrict__ Wv, const float* __restrict__ bq,
                       const float* __restrict__ bk, const float* __restrict__ bv,
                       __bf16* __restrict__ wqkvT, float* __restrict__ bqkv) {
  int i = blockIdx.x * 256 + threadIdx.x;
  if (i < QKVW * CCH) {
    int n = i / CCH, kk = i % CCH;
    float w = (n < 32) ? Wq[kk * 32 + n]
            : (n < 64) ? Wk[kk * 32 + (n - 32)]
                       : Wv[kk * CCH + (n - 64)];
    wqkvT[i] = f2bf(w);
  }
  if (i < QKVW)
    bqkv[i] = (i < 32) ? bq[i] : (i < 64) ? bk[i - 32] : bv[i - 64];
}

__global__ void k_x2(const float* __restrict__ x, const float* __restrict__ tfeat,
                     float* __restrict__ x2f, __bf16* __restrict__ x2b) {
  size_t i = (size_t)blockIdx.x * 256 + threadIdx.x;
  int c = (int)(i & (CCH - 1));
  int b = (int)(i >> 20);
  float v = x[i] + tfeat[b * CCH + c];
  x2f[i] = v;
  x2b[i] = f2bf(v);
}

__global__ __launch_bounds__(128, 2) void k_qkv(
    const __bf16* __restrict__ x2b, const __bf16* __restrict__ wqkvT,
    const float* __restrict__ bqkv, __bf16* __restrict__ qk,
    __bf16* __restrict__ vT) {
  const int mt = blockIdx.x;
  const int nt = blockIdx.y;
  const int wave = threadIdx.x >> 5;
  const int lane = threadIdx.x & 31;
  const int half = lane >> 4, l16 = lane & 15;
  const int row = mt * 64 + wave * 16 + l16;
  const int colbase = nt * 64;

  v8f acc[4];
  #pragma unroll
  for (int t = 0; t < 4; ++t)
    #pragma unroll
    for (int j = 0; j < 8; ++j) acc[t][j] = 0.f;

  for (int k0 = 0; k0 < CCH; k0 += 32) {
    v16bf a;
    const __bf16* ar = x2b + (size_t)row * CCH + k0;
    #pragma unroll
    for (int r = 0; r < 8; ++r) {
      int kk = a_kpair(r, half);
      a[2 * r]     = ar[kk];
      a[2 * r + 1] = ar[kk + 1];
    }
    #pragma unroll
    for (int t = 0; t < 4; ++t) {
      int col = colbase + t * 16 + l16;
      const __bf16* br = wqkvT + (size_t)col * CCH + k0 + 16 * half;
      v16bf b = *reinterpret_cast<const v16bf*>(br);
      acc[t] = __builtin_amdgcn_wmma_f32_16x16x32_bf16(
          false, a, false, b, (short)0, acc[t], false, false);
    }
  }

  const int bidx = mt >> 6;
  #pragma unroll
  for (int t = 0; t < 4; ++t) {
    int col = colbase + t * 16 + l16;
    float bias = bqkv[col];
    #pragma unroll
    for (int j = 0; j < 8; ++j) {
      int r = mt * 64 + wave * 16 + j + 8 * half;
      __bf16 val = f2bf(acc[t][j] + bias);
      if (nt == 0) {
        qk[(size_t)r * QKW + col] = val;
      } else {
        int vc = col - 64;
        int tok = r & (NTOK - 1);
        vT[((size_t)bidx * CCH + vc) * NTOK + tok] = val;
      }
    }
  }
}

__global__ __launch_bounds__(256, 1) void k_attn(
    const __bf16* __restrict__ qk, const __bf16* __restrict__ vT,
    const float* __restrict__ x2f, const float* __restrict__ gamma,
    float* __restrict__ out) {
  const int b    = blockIdx.x;
  const int qt   = blockIdx.y;
  const int wave = threadIdx.x >> 5;
  const int lane = threadIdx.x & 31;
  const int half = lane >> 4, l16 = lane & 15;
  const int rb   = wave & 3;
  const int ch0  = (wave >> 2) * 128;
  const size_t tokbase = (size_t)b * NTOK;
  const int qbase = (int)tokbase + qt * 64 + rb * 16;
  const float scale = 0.17677669529663687f;
  const __bf16* vbat = vT + (size_t)b * CCH * NTOK;

  __shared__ __bf16 lds_p[8][16 * KVT];
  __bf16* pbuf = lds_p[wave];

  v16bf qa;
  {
    const __bf16* qp = qk + (size_t)(qbase + l16) * QKW;
    #pragma unroll
    for (int r = 0; r < 8; ++r) {
      int kk = a_kpair(r, half);
      qa[2 * r]     = qp[kk];
      qa[2 * r + 1] = qp[kk + 1];
    }
  }

  float mrow[8], lrow[8];
  v8f acc[8];
  #pragma unroll
  for (int j = 0; j < 8; ++j) { mrow[j] = -3.0e38f; lrow[j] = 0.f; }
  #pragma unroll
  for (int t = 0; t < 8; ++t)
    #pragma unroll
    for (int j = 0; j < 8; ++j) acc[t][j] = 0.f;

  v8f zf;
  #pragma unroll
  for (int j = 0; j < 8; ++j) zf[j] = 0.f;

  for (int kv = 0; kv < NTOK; kv += KVT) {
    if (kv + KVT < NTOK) {
      __builtin_prefetch(qk + (tokbase + kv + KVT + lane) * QKW, 0, 1);
      #pragma unroll
      for (int t = 0; t < 8; ++t)
        __builtin_prefetch(
            vbat + (size_t)(ch0 + t * 16 + l16) * NTOK + kv + KVT + 32 * half,
            0, 1);
    }

    v8f s[4];
    #pragma unroll
    for (int nt = 0; nt < 4; ++nt) {
      const __bf16* kp =
          qk + (size_t)(tokbase + kv + nt * 16 + l16) * QKW + DQK + 16 * half;
      v16bf kb = *reinterpret_cast<const v16bf*>(kp);
      s[nt] = __builtin_amdgcn_wmma_f32_16x16x32_bf16(
          false, qa, false, kb, (short)0, zf, false, false);
    }

    float corr[8];
    #pragma unroll
    for (int j = 0; j < 8; ++j) {
      float v = fmaxf(fmaxf(s[0][j], s[1][j]), fmaxf(s[2][j], s[3][j]));
      v = rmax16(v) * scale;
      float mnew = fmaxf(mrow[j], v);
      corr[j] = __expf(mrow[j] - mnew);
      mrow[j] = mnew;
      lrow[j] *= corr[j];
    }
    float psum[8];
    #pragma unroll
    for (int j = 0; j < 8; ++j) psum[j] = 0.f;
    #pragma unroll
    for (int nt = 0; nt < 4; ++nt)
      #pragma unroll
      for (int j = 0; j < 8; ++j) {
        float p = __expf(s[nt][j] * scale - mrow[j]);
        pbuf[(j + 8 * half) * KVT + nt * 16 + l16] = f2bf(p);
        psum[j] += p;
      }
    #pragma unroll
    for (int j = 0; j < 8; ++j) lrow[j] += rsum16(psum[j]);

    #pragma unroll
    for (int t = 0; t < 8; ++t)
      #pragma unroll
      for (int j = 0; j < 8; ++j) acc[t][j] *= corr[j];

    v16bf pa[2];
    #pragma unroll
    for (int kt = 0; kt < 2; ++kt)
      #pragma unroll
      for (int r = 0; r < 8; ++r) {
        int kk = kt * 32 + a_kpair(r, half);
        pa[kt][2 * r]     = pbuf[l16 * KVT + kk];
        pa[kt][2 * r + 1] = pbuf[l16 * KVT + kk + 1];
      }

    #pragma unroll
    for (int t = 0; t < 8; ++t) {
      const __bf16* vp =
          vbat + (size_t)(ch0 + t * 16 + l16) * NTOK + kv + 16 * half;
      #pragma unroll
      for (int kt = 0; kt < 2; ++kt) {
        v16bf vb = *reinterpret_cast<const v16bf*>(vp + kt * 32);
        acc[t] = __builtin_amdgcn_wmma_f32_16x16x32_bf16(
            false, pa[kt], false, vb, (short)0, acc[t], false, false);
      }
    }
  }

  const float g = gamma[0];
  #pragma unroll
  for (int j = 0; j < 8; ++j) {
    float inv = 1.0f / lrow[j];
    int tok = qbase + j + 8 * half;
    const float* xr = x2f + (size_t)tok * CCH;
    float* orow = out + (size_t)tok * CCH;
    #pragma unroll
    for (int t = 0; t < 8; ++t) {
      int col = ch0 + t * 16 + l16;
      orow[col] = g * (acc[t][j] * inv) + xr[col];
    }
  }
}

extern "C" void kernel_launch(void* const* d_in, const int* in_sizes, int n_in,
                              void* d_out, int out_size, void* d_ws, size_t ws_size,
                              hipStream_t stream) {
  (void)in_sizes; (void)n_in; (void)out_size; (void)ws_size;
  const float* x       = (const float*)d_in[0];
  const float* terrain = (const float*)d_in[1];
  const float* Wq      = (const float*)d_in[2];
  const float* bq      = (const float*)d_in[3];
  const float* Wk      = (const float*)d_in[4];
  const float* bk      = (const float*)d_in[5];
  const float* Wv      = (const float*)d_in[6];
  const float* bv      = (const float*)d_in[7];
  const float* Wt      = (const float*)d_in[8];
  const float* bt      = (const float*)d_in[9];
  const float* gamma   = (const float*)d_in[10];
  float* out = (float*)d_out;

  char* ws = (char*)d_ws;
  float*  tfeat  = (float*) (ws + 0);
  float*  bqkv   = (float*) (ws + 4096);
  __bf16* wqkvT  = (__bf16*)(ws + 8192);
  float*  x2f    = (float*) (ws + 262144);
  __bf16* x2b    = (__bf16*)(ws + 17039360);
  __bf16* qk     = (__bf16*)(ws + 25427968);
  __bf16* vT     = (__bf16*)(ws + 27525120);

  k_tfeat<<<BATCH, CCH, 0, stream>>>(terrain, Wt, bt, tfeat);
  k_pack<<<(QKVW * CCH + 255) / 256, 256, 0, stream>>>(Wq, Wk, Wv, bq, bk, bv,
                                                       wqkvT, bqkv);
  k_x2<<<(BATCH * NTOK * CCH) / 256, 256, 0, stream>>>(x, tfeat, x2f, x2b);
  k_qkv<<<dim3(BATCH * NTOK / 64, QKVW / 64), 128, 0, stream>>>(x2b, wqkvT,
                                                                bqkv, qk, vT);
  k_attn<<<dim3(BATCH, NTOK / 64), 256, 0, stream>>>(qk, vT, x2f, gamma, out);
}
